// QuantizedLayer_60026462929270
// MI455X (gfx1250) — matrix operand order repaired and checked by reading, not compiled, not run on hardware
//
#include <hip/hip_runtime.h>

typedef __attribute__((ext_vector_type(16))) _Float16 v16h;
typedef __attribute__((ext_vector_type(8)))  _Float16 v8h;
typedef __attribute__((ext_vector_type(4)))  _Float16 v4h;
typedef __attribute__((ext_vector_type(8)))  float    v8f;
typedef __attribute__((ext_vector_type(4)))  float    v4f;
typedef __attribute__((ext_vector_type(4)))  int      v4i;

#define BM 128
#define BN 128
#define BK 32
#define THREADS 256

// GEMM: out[m,n] = (sum_k x[m,k] * q[n,k]) / scale[n] + bias[n]
// M=512, K=4096, N=11008.  A (x) and B (q) staged through double-buffered LDS
// as f16; compute with V_WMMA_F32_16X16X32_F16 (f32 accumulate).
__global__ __launch_bounds__(THREADS, 1)
void qlinear_wmma_kernel(const float* __restrict__ X, const int* __restrict__ Q,
                         const float* __restrict__ Scale, const float* __restrict__ Bias,
                         float* __restrict__ Out, int M, int N, int K)
{
    __shared__ alignas(32) _Float16 lA[2][BM * BK];   // 8 KB each buffer
    __shared__ alignas(32) _Float16 lB[2][BN * BK];   // stored as [n][k] (q is already n-major)

    const int t  = threadIdx.x;
    const int m0 = blockIdx.x * BM;   // M fastest-varying -> 4 blocks sharing a Q stripe run adjacently (L2 reuse)
    const int n0 = blockIdx.y * BN;

    const int wave = t >> 5;
    const int lane = t & 31;
    const int wm = wave & 1;          // wave grid 2(M) x 4(N)
    const int wn = wave >> 1;
    const int lm = lane & 15;
    const int lh = lane >> 4;

    v8f acc[4][2];
    #pragma unroll
    for (int i = 0; i < 4; ++i)
        #pragma unroll
        for (int j = 0; j < 2; ++j)
            acc[i][j] = (v8f){0.f, 0.f, 0.f, 0.f, 0.f, 0.f, 0.f, 0.f};

    v4f aR[4];   // staging: 4 x float4 per thread (128x32 f32 tile / 256 threads)
    v4i bR[4];   // staging: 4 x int4   per thread

    const int KT = K / BK;

    // ---- prologue: fetch + convert + stage tile 0 ----
    #pragma unroll
    for (int i = 0; i < 4; ++i) {
        int v = t + i * THREADS;
        int row = v >> 3;              // 8 vec4 per row of 32
        int c4  = (v & 7) << 2;
        aR[i] = *(const v4f*)(X + (size_t)(m0 + row) * K + c4);
        bR[i] = *(const v4i*)(Q + (size_t)(n0 + row) * K + c4);
    }
    #pragma unroll
    for (int i = 0; i < 4; ++i) {
        int v = t + i * THREADS;
        int row = v >> 3;
        int c4  = (v & 7) << 2;
        v4h ha, hb;
        #pragma unroll
        for (int e = 0; e < 4; ++e) {
            ha[e] = (_Float16)aR[i][e];
            hb[e] = (_Float16)bR[i][e];   // int8 values are exact in f16
        }
        *(v4h*)&lA[0][row * BK + c4] = ha;
        *(v4h*)&lB[0][row * BK + c4] = hb;
    }
    __syncthreads();

    int buf = 0;
    for (int kt = 0; kt < KT; ++kt) {
        // issue next tile's global loads before compute (latency overlap)
        if (kt + 1 < KT) {
            const int kb = (kt + 1) * BK;
            #pragma unroll
            for (int i = 0; i < 4; ++i) {
                int v = t + i * THREADS;
                int row = v >> 3;
                int c4  = (v & 7) << 2;
                aR[i] = *(const v4f*)(X + (size_t)(m0 + row) * K + kb + c4);
                bR[i] = *(const v4i*)(Q + (size_t)(n0 + row) * K + kb + c4);
            }
        }

        // ---- compute on current buffer ----
        const _Float16* pA = &lA[buf][0];
        const _Float16* pB = &lB[buf][0];

        // B fragment: 32x16 f16, lane = column lm, K = lh*16 .. +15 -> one contiguous 32B chunk
        v16h bfr[2];
        #pragma unroll
        for (int nt = 0; nt < 2; ++nt) {
            int c = wn * 32 + nt * 16 + lm;
            bfr[nt] = *(const v16h*)(pB + c * BK + lh * 16);
        }

        #pragma unroll
        for (int mt = 0; mt < 4; ++mt) {
            // A fragment: 16x32 f16, row r, chunks K = lh*16 and lh*16 + 8
            int r = wm * 64 + mt * 16 + lm;
            v8h lo = *(const v8h*)(pA + r * BK + lh * 16);
            v8h hi = *(const v8h*)(pA + r * BK + lh * 16 + 8);
            v16h afr;
            #pragma unroll
            for (int e = 0; e < 8; ++e) { afr[e] = lo[e]; afr[e + 8] = hi[e]; }

            acc[mt][0] = __builtin_amdgcn_wmma_f32_16x16x32_f16(
                false, afr, false, bfr[0], (short)0, acc[mt][0], false, false);
            acc[mt][1] = __builtin_amdgcn_wmma_f32_16x16x32_f16(
                false, afr, false, bfr[1], (short)0, acc[mt][1], false, false);
        }

        // ---- convert + stage next tile into alternate buffer ----
        if (kt + 1 < KT) {
            const int nb = buf ^ 1;
            #pragma unroll
            for (int i = 0; i < 4; ++i) {
                int v = t + i * THREADS;
                int row = v >> 3;
                int c4  = (v & 7) << 2;
                v4h ha, hb;
                #pragma unroll
                for (int e = 0; e < 4; ++e) {
                    ha[e] = (_Float16)aR[i][e];
                    hb[e] = (_Float16)bR[i][e];
                }
                *(v4h*)&lA[nb][row * BK + c4] = ha;
                *(v4h*)&lB[nb][row * BK + c4] = hb;
            }
            __syncthreads();
            buf = nb;
        }
    }

    // ---- epilogue: dequant scale + bias, write f32 ----
    #pragma unroll
    for (int nt = 0; nt < 2; ++nt) {
        int n = n0 + wn * 32 + nt * 16 + lm;
        float s = 1.0f / Scale[n];
        float b = Bias[n];
        #pragma unroll
        for (int mt = 0; mt < 4; ++mt) {
            #pragma unroll
            for (int r = 0; r < 8; ++r) {
                int m = m0 + wm * 64 + mt * 16 + lh * 8 + r;   // f32 C/D VGPR layout
                Out[(size_t)m * N + n] = acc[mt][nt][r] * s + b;
            }
        }
    }
}

extern "C" void kernel_launch(void* const* d_in, const int* in_sizes, int n_in,
                              void* d_out, int out_size, void* d_ws, size_t ws_size,
                              hipStream_t stream) {
    const float* X  = (const float*)d_in[0];   // (4,128,4096) f32
    const int*   Q  = (const int*)d_in[1];     // (11008,4096) int32 in [-127,127]
    const float* S  = (const float*)d_in[2];   // (11008,1)
    const float* B  = (const float*)d_in[3];   // (11008,)
    float* O = (float*)d_out;                  // (4,128,11008) f32

    const int K = 4096;
    const int M = in_sizes[0] / K;             // 512
    const int N = in_sizes[1] / K;             // 11008

    dim3 grid(M / BM, N / BN);                 // (4, 86); M fastest for Q-stripe L2 reuse
    qlinear_wmma_kernel<<<grid, THREADS, 0, stream>>>(X, Q, S, B, O, M, N, K);
}
